// DiaM_64029372449086
// MI455X (gfx1250) — matrix operand order repaired and checked by reading, not compiled, not run on hardware
//
#include <hip/hip_runtime.h>
#include <hip/hip_bf16.h>
#include <math.h>

// ---------------------------------------------------------------------------
// MHA forward for MI455X (gfx1250, wave32, WMMA bf16 16x16x32 + TDM).
//  K1: fused QKV projection (f32 x, f32 W -> bf16 Q/K/V in [B,H,N,d])
//  K2: per-(b,h) causal attention, flash-style two-pass softmax; K/Q tiles
//      staged via TENSOR_LOAD_TO_LDS with hardware row padding.
//  K3: output projection; whole bf16 A panel staged once per WG via TDM.
// Workspace layout (bf16/u16): Qh | Kh | Vh | attn  = 4 * 33,554,432 elems
//                              = 268,435,456 bytes required in d_ws.
// ---------------------------------------------------------------------------

typedef unsigned short u16;
typedef __bf16 bf16_t;
typedef bf16_t v16bf __attribute__((ext_vector_type(16)));
typedef float  v8f   __attribute__((ext_vector_type(8)));
typedef unsigned int v4u  __attribute__((ext_vector_type(4)));
typedef unsigned int u32x4 __attribute__((ext_vector_type(4)));
typedef int i32x4 __attribute__((ext_vector_type(4)));
typedef int i32x8 __attribute__((ext_vector_type(8)));

union Frag16 { v4u q[2]; v16bf v; };   // 32B: one WMMA A/B operand (8 VGPRs)

__device__ __forceinline__ u16 f2bf(float f) {
  unsigned u = __float_as_uint(f);
  u += 0x7FFFu + ((u >> 16) & 1u);     // round-to-nearest-even
  return (u16)(u >> 16);
}

static constexpr int Bb = 256, Nn = 256, Cc = 512, Hh = 8, Dd = 64;
static constexpr int Mrows = Bb * Nn;  // 65536

// ---------------------------------------------------------------------------
// Tensor Data Mover: 2D tile (row_dwords x nrows) global -> LDS, optional
// per-row LDS padding. D# packing per CDNA5 ISA.
// pad_interval code: pad after 2^(code+1) DWORDs; pad_amount code: (code+1) DW.
// ---------------------------------------------------------------------------
#if __has_builtin(__builtin_amdgcn_tensor_load_to_lds)
#define USE_TDM 1
__device__ __forceinline__ void tdm_load_2d(const void* gptr, unsigned lds_off,
                                            unsigned row_dwords, unsigned nrows,
                                            unsigned stride_dwords,
                                            unsigned pad_int_code,
                                            unsigned pad_amt_code) {
  unsigned long long ga = (unsigned long long)(uintptr_t)gptr;
  u32x4 g0;
  g0.x = 1u;                                         // count=1 (valid), user
  g0.y = lds_off;                                    // LDS byte address
  g0.z = (unsigned)(ga & 0xFFFFFFFFu);               // global addr [31:0]
  g0.w = (unsigned)((ga >> 32) & 0x01FFFFFFu) | (2u << 30);  // [56:32]|type=2
  i32x8 g1;
  g1[0] = (int)((2u << 16) |                         // data_size = 4B
                (1u << 20) |                         // pad_enable
                (pad_int_code << 22) | (pad_amt_code << 25));
  g1[1] = (int)((row_dwords & 0xFFFFu) << 16);       // tensor_dim0 [15:0]
  g1[2] = (int)(((row_dwords >> 16) & 0xFFFFu) |     // tensor_dim0 [31:16]
                ((nrows & 0xFFFFu) << 16));          // tensor_dim1 [15:0]
  g1[3] = (int)(((nrows >> 16) & 0xFFFFu) |          // tensor_dim1 [31:16]
                ((row_dwords & 0xFFFFu) << 16));     // tile_dim0
  g1[4] = (int)(nrows & 0xFFFFu);                    // tile_dim1 (tile_dim2=0)
  g1[5] = (int)stride_dwords;                        // tensor_dim0_stride lo
  g1[6] = 0;
  g1[7] = 0;
  i32x4 z4 = {0, 0, 0, 0};
#if __has_include(<hip/amd_detail/amd_gfx1250_TDM.h>)
  i32x8 z8 = {0, 0, 0, 0, 0, 0, 0, 0};               // clang-23 6-arg variant
  __builtin_amdgcn_tensor_load_to_lds(g0, g1, z4, z4, z8, 0);
#else
  __builtin_amdgcn_tensor_load_to_lds(g0, g1, z4, z4, 0);  // ROCm 7.2 5-arg
#endif
}
__device__ __forceinline__ unsigned lds_off_of(const void* p) {
  return (unsigned)(uintptr_t)p;  // flat shared addr: low 32 bits = LDS byte
}
#endif

// ---------------------------------------------------------------------------
// Kernel 1: fused QKV projection.
// grid (Mrows/64, Cc/128), block 256 (8 waves). Wave w: row-tile w>>1,
// col half (w&1)*64 (4 col sub-tiles) for each of the 3 weights.
// ---------------------------------------------------------------------------
__global__ __launch_bounds__(256)
void qkv_proj_kernel(const float* __restrict__ x,
                     const float* __restrict__ Wq, const float* __restrict__ bq,
                     const float* __restrict__ Wk, const float* __restrict__ bk,
                     const float* __restrict__ Wv, const float* __restrict__ bv,
                     u16* __restrict__ Qh, u16* __restrict__ Kh,
                     u16* __restrict__ Vh) {
  __shared__ u16 Atile[64 * 40];
  __shared__ u16 Btile[3][128 * 40];

  const int t  = threadIdx.x;
  const int r0 = blockIdx.x * 64;
  const int c0 = blockIdx.y * 128;
  const int w  = t >> 5, l = t & 31;
  const int lo = l & 15, hi = l >> 4;
  const int rw = (w >> 1) * 16;
  const int cwBase = (w & 1) * 64;
  const float* Ws[3] = {Wq, Wk, Wv};

  v8f acc[3][4];
#pragma unroll
  for (int j = 0; j < 3; ++j)
#pragma unroll
    for (int ct = 0; ct < 4; ++ct) acc[j][ct] = {};

  for (int kc = 0; kc < Cc; kc += 32) {
    __syncthreads();
    {  // stage A: 64 rows x 32 k of x, f32 -> bf16
      const int row = t >> 2, k = (t & 3) * 8;
      const float* src = x + (size_t)(r0 + row) * Cc + kc + k;
#pragma unroll
      for (int j = 0; j < 8; ++j) Atile[row * 40 + k + j] = f2bf(src[j]);
      if (kc + 32 < Cc) __builtin_prefetch(src + 32, 0, 1);  // global_prefetch_b8
    }
    // stage B (3 weights) transposed: W[k][c] -> Btile[c][k]
#pragma unroll
    for (int j3 = 0; j3 < 3; ++j3) {
      const float* Wf = Ws[j3];
#pragma unroll
      for (int i = 0; i < 16; ++i) {
        int f = t * 16 + i;            // 32*128 floats, c fastest
        int k = f >> 7, c = f & 127;
        Btile[j3][c * 40 + k] = f2bf(Wf[(size_t)(kc + k) * Cc + c0 + c]);
      }
    }
    __syncthreads();

    Frag16 a;  // A 16x32, k order used here (same for B): 16 consecutive K at kb=hi?16:0
    {
      const int row = rw + lo, kb = hi ? 16 : 0;
      a.q[0] = *reinterpret_cast<const v4u*>(&Atile[row * 40 + kb]);
      a.q[1] = *reinterpret_cast<const v4u*>(&Atile[row * 40 + kb + 8]);
    }
#pragma unroll
    for (int j3 = 0; j3 < 3; ++j3) {
#pragma unroll
      for (int ct = 0; ct < 4; ++ct) {
        Frag16 bfr;  // B 32x16: lane=col, 16 consecutive k at kb2=hi?16:0
        const int col = cwBase + ct * 16 + lo, kb2 = hi ? 16 : 0;
        bfr.q[0] = *reinterpret_cast<const v4u*>(&Btile[j3][col * 40 + kb2]);
        bfr.q[1] = *reinterpret_cast<const v4u*>(&Btile[j3][col * 40 + kb2 + 8]);
        acc[j3][ct] = __builtin_amdgcn_wmma_f32_16x16x32_bf16(
            false, a.v, false, bfr.v, (short)0, acc[j3][ct], false, false);
      }
    }
  }

  const float* bias[3] = {bq, bk, bv};
  u16* outs[3] = {Qh, Kh, Vh};
#pragma unroll
  for (int j3 = 0; j3 < 3; ++j3) {
#pragma unroll
    for (int ct = 0; ct < 4; ++ct) {
      const int cg = c0 + cwBase + ct * 16 + lo;
      const float bb = bias[j3][cg];
      const int h = cg >> 6, dd = cg & 63;
#pragma unroll
      for (int v = 0; v < 8; ++v) {     // C layout: lane=col, VGPR v -> M=v(+8)
        const int rg = r0 + rw + v + hi * 8;
        const int b = rg >> 8, n = rg & 255;
        outs[j3][((size_t)((b * Hh + h) * Nn + n)) * Dd + dd] =
            f2bf(acc[j3][ct][v] + bb);
      }
    }
  }
}

// ---------------------------------------------------------------------------
// Kernel 2: causal attention for one (b,h), 128-query block per workgroup.
// grid (B*H, 2), block 256 (8 waves; wave owns 16 query rows).
// K[256][72] and Q[128][72] staged by TDM with hardware pad (32 DW rows +
// 4 DW pad -> stride 72 bf16); V transposed manually into Vt[64][264].
// ---------------------------------------------------------------------------
__global__ __launch_bounds__(256)
void attn_kernel(const u16* __restrict__ Qh, const u16* __restrict__ Kh,
                 const u16* __restrict__ Vh, u16* __restrict__ attnOut) {
  __shared__ u16 Ks[256 * 72];
  __shared__ u16 Vt[64 * 264];
  __shared__ u16 Qs[128 * 72];
  __shared__ u16 Pst[8 * 16 * 40];

  const int t  = threadIdx.x;
  const int bh = blockIdx.x;
  const int b = bh >> 3, h = bh & 7;
  const int q0 = blockIdx.y * 128;
  const size_t head = (size_t)bh * (Nn * Dd);
  const int w = t >> 5, l = t & 31, lo = l & 15, hi = l >> 4;

#if USE_TDM
  if (w == 0) {  // one wave issues both DMAs; TDM ignores EXEC
    tdm_load_2d(Kh + head, lds_off_of(Ks), /*row_dw=*/32, /*rows=*/256,
                /*stride_dw=*/32, /*pad@32DW=*/4, /*pad 4DW=*/3);
    tdm_load_2d(Qh + head + (size_t)q0 * 64, lds_off_of(Qs), 32, 128, 32, 4, 3);
  }
#else
#pragma unroll
  for (int i = 0; i < 8; ++i) {        // K: 2048 x v4u
    int cid = t * 8 + i, key = cid >> 3, part = (cid & 7) * 8;
    *reinterpret_cast<v4u*>(&Ks[key * 72 + part]) =
        *reinterpret_cast<const v4u*>(&Kh[head + key * 64 + part]);
  }
#pragma unroll
  for (int i = 0; i < 4; ++i) {        // Q block: 1024 x v4u
    int cid = t * 4 + i, row = cid >> 3, part = (cid & 7) * 8;
    *reinterpret_cast<v4u*>(&Qs[row * 72 + part]) =
        *reinterpret_cast<const v4u*>(&Qh[head + (size_t)(q0 + row) * 64 + part]);
  }
#endif
#pragma unroll
  for (int i = 0; i < 8; ++i) {        // V transposed into Vt[dd][key]
    int cid = t * 8 + i, key = cid >> 3, part = (cid & 7) * 8;
    union { v4u q; u16 s[8]; } u;
    u.q = *reinterpret_cast<const v4u*>(&Vh[head + key * 64 + part]);
#pragma unroll
    for (int j = 0; j < 8; ++j) Vt[(part + j) * 264 + key] = u.s[j];
  }
#if USE_TDM
  __builtin_amdgcn_s_wait_tensorcnt(0);  // no-op for non-issuing waves
#endif
  __syncthreads();

  const int qr0 = w * 16;
  const int gq0 = q0 + qr0;
  const int ktEnd = (q0 + 127) >> 4;   // causal: skip fully-masked key tiles
  const int kbA = hi ? 16 : 0, kbB = hi ? 16 : 0;

  Frag16 aq0, aq1;                     // Q fragments for k-chunks 0 and 32
  {
    const int row = qr0 + lo;
    aq0.q[0] = *reinterpret_cast<const v4u*>(&Qs[row * 72 + kbA]);
    aq0.q[1] = *reinterpret_cast<const v4u*>(&Qs[row * 72 + kbA + 8]);
    aq1.q[0] = *reinterpret_cast<const v4u*>(&Qs[row * 72 + 32 + kbA]);
    aq1.q[1] = *reinterpret_cast<const v4u*>(&Qs[row * 72 + 32 + kbA + 8]);
  }

  // ---- pass 1: row max ----
  float m[8];
#pragma unroll
  for (int v = 0; v < 8; ++v) m[v] = -3.0e38f;
  for (int kt = 0; kt <= ktEnd; ++kt) {
    const int key = kt * 16 + lo;
    Frag16 bk0, bk1;
    bk0.q[0] = *reinterpret_cast<const v4u*>(&Ks[key * 72 + kbB]);
    bk0.q[1] = *reinterpret_cast<const v4u*>(&Ks[key * 72 + kbB + 8]);
    bk1.q[0] = *reinterpret_cast<const v4u*>(&Ks[key * 72 + 32 + kbB]);
    bk1.q[1] = *reinterpret_cast<const v4u*>(&Ks[key * 72 + 32 + kbB + 8]);
    v8f acc = {};
    acc = __builtin_amdgcn_wmma_f32_16x16x32_bf16(false, aq0.v, false, bk0.v,
                                                  (short)0, acc, false, false);
    acc = __builtin_amdgcn_wmma_f32_16x16x32_bf16(false, aq1.v, false, bk1.v,
                                                  (short)0, acc, false, false);
#pragma unroll
    for (int v = 0; v < 8; ++v) {
      const int qg = gq0 + v + hi * 8;
      // reference adds mask BEFORE dividing by sqrt(d)
      const float s = (key <= qg) ? acc[v] * 0.125f : (acc[v] - 1.0e5f) * 0.125f;
      m[v] = fmaxf(m[v], s);
    }
  }
#pragma unroll
  for (int off = 1; off < 16; off <<= 1)
#pragma unroll
    for (int v = 0; v < 8; ++v) m[v] = fmaxf(m[v], __shfl_xor(m[v], off, 32));

  // ---- pass 2: exp, P.V accumulate (k=32 per WMMA -> 2 key tiles) ----
  float sum[8];
#pragma unroll
  for (int v = 0; v < 8; ++v) sum[v] = 0.0f;
  v8f o[4];
#pragma unroll
  for (int dt = 0; dt < 4; ++dt) o[dt] = {};

  u16* pbase = &Pst[w * 640];
  for (int kt2 = 0; kt2 <= ktEnd; kt2 += 2) {
#pragma unroll
    for (int sub = 0; sub < 2; ++sub) {
      const int kt = kt2 + sub;
      const int key = kt * 16 + lo;
      Frag16 bk0, bk1;
      bk0.q[0] = *reinterpret_cast<const v4u*>(&Ks[key * 72 + kbB]);
      bk0.q[1] = *reinterpret_cast<const v4u*>(&Ks[key * 72 + kbB + 8]);
      bk1.q[0] = *reinterpret_cast<const v4u*>(&Ks[key * 72 + 32 + kbB]);
      bk1.q[1] = *reinterpret_cast<const v4u*>(&Ks[key * 72 + 32 + kbB + 8]);
      v8f acc = {};
      acc = __builtin_amdgcn_wmma_f32_16x16x32_bf16(false, aq0.v, false, bk0.v,
                                                    (short)0, acc, false, false);
      acc = __builtin_amdgcn_wmma_f32_16x16x32_bf16(false, aq1.v, false, bk1.v,
                                                    (short)0, acc, false, false);
#pragma unroll
      for (int v = 0; v < 8; ++v) {
        const int qg = gq0 + v + hi * 8;
        const float s = (key <= qg) ? acc[v] * 0.125f : (acc[v] - 1.0e5f) * 0.125f;
        const float p = __expf(s - m[v]);
        sum[v] += p;
        // C-layout -> P staging row (v+hi*8), col = key within 32-chunk
        pbase[(v + hi * 8) * 40 + sub * 16 + lo] = f2bf(p);
      }
    }
    __builtin_amdgcn_wave_barrier();   // LDS is in-order per wave on CDNA5
    Frag16 ap;                         // P as A fragment (16x32)
    ap.q[0] = *reinterpret_cast<const v4u*>(&pbase[lo * 40 + kbA]);
    ap.q[1] = *reinterpret_cast<const v4u*>(&pbase[lo * 40 + kbA + 8]);
#pragma unroll
    for (int dt = 0; dt < 4; ++dt) {
      Frag16 bv;                       // V^T as B: col = dd, k = key
      const int dd = dt * 16 + lo;
      bv.q[0] = *reinterpret_cast<const v4u*>(&Vt[dd * 264 + kt2 * 16 + kbB]);
      bv.q[1] = *reinterpret_cast<const v4u*>(&Vt[dd * 264 + kt2 * 16 + kbB + 8]);
      o[dt] = __builtin_amdgcn_wmma_f32_16x16x32_bf16(
          false, ap.v, false, bv.v, (short)0, o[dt], false, false);
    }
    __builtin_amdgcn_wave_barrier();
  }
#pragma unroll
  for (int off = 1; off < 16; off <<= 1)
#pragma unroll
    for (int v = 0; v < 8; ++v) sum[v] += __shfl_xor(sum[v], off, 32);

  // write attn in [B, N, C] (C = h*64 + dd), bf16
#pragma unroll
  for (int dt = 0; dt < 4; ++dt) {
    const int dd = dt * 16 + lo;
#pragma unroll
    for (int v = 0; v < 8; ++v) {
      const int nrow = gq0 + v + hi * 8;
      attnOut[((size_t)(b * Nn + nrow)) * Cc + h * 64 + dd] =
          f2bf(o[dt][v] / sum[v]);
    }
  }
}

// ---------------------------------------------------------------------------
// Kernel 3: output projection. grid (Mrows/64, Cc/128), block 256.
// Whole 64x512 bf16 A panel staged ONCE per workgroup by TDM (256 DW rows +
// 4 DW pad -> stride 520 bf16); only the Wo tile is staged per k-step.
// ---------------------------------------------------------------------------
__global__ __launch_bounds__(256)
void out_proj_kernel(const u16* __restrict__ attnIn,
                     const float* __restrict__ Wo,
                     const float* __restrict__ bo, float* __restrict__ out) {
  __shared__ u16 Apanel[64 * 520];
  __shared__ u16 Btile[128 * 40];

  const int t  = threadIdx.x;
  const int r0 = blockIdx.x * 64, c0 = blockIdx.y * 128;
  const int w  = t >> 5, l = t & 31, lo = l & 15, hi = l >> 4;
  const int rw = (w >> 1) * 16, cwBase = (w & 1) * 64;

#if USE_TDM
  if (w == 0)
    tdm_load_2d(attnIn + (size_t)r0 * Cc, lds_off_of(Apanel),
                /*row_dw=*/256, /*rows=*/64, /*stride_dw=*/256,
                /*pad@256DW=*/7, /*pad 4DW=*/3);
  __builtin_amdgcn_s_wait_tensorcnt(0);
#else
#pragma unroll
  for (int i = 0; i < 16; ++i) {       // 4096 x v4u panel copy
    int cid = t * 16 + i, row = cid >> 6, part = (cid & 63) * 8;
    *reinterpret_cast<v4u*>(&Apanel[row * 520 + part]) =
        *reinterpret_cast<const v4u*>(&attnIn[(size_t)(r0 + row) * Cc + part]);
  }
#endif

  v8f acc[4];
#pragma unroll
  for (int ct = 0; ct < 4; ++ct) acc[ct] = {};

  for (int kc = 0; kc < Cc; kc += 32) {
    __syncthreads();
#pragma unroll
    for (int i = 0; i < 16; ++i) {     // Wo transposed, f32 -> bf16
      int f = t * 16 + i;
      int k = f >> 7, c = f & 127;
      Btile[c * 40 + k] = f2bf(Wo[(size_t)(kc + k) * Cc + c0 + c]);
    }
    __syncthreads();

    Frag16 a;
    const int row = rw + lo, kb = hi ? 16 : 0;
    a.q[0] = *reinterpret_cast<const v4u*>(&Apanel[row * 520 + kc + kb]);
    a.q[1] = *reinterpret_cast<const v4u*>(&Apanel[row * 520 + kc + kb + 8]);
#pragma unroll
    for (int ct = 0; ct < 4; ++ct) {
      Frag16 bfr;
      const int col = cwBase + ct * 16 + lo, kb2 = hi ? 16 : 0;
      bfr.q[0] = *reinterpret_cast<const v4u*>(&Btile[col * 40 + kb2]);
      bfr.q[1] = *reinterpret_cast<const v4u*>(&Btile[col * 40 + kb2 + 8]);
      acc[ct] = __builtin_amdgcn_wmma_f32_16x16x32_bf16(
          false, a.v, false, bfr.v, (short)0, acc[ct], false, false);
    }
  }

#pragma unroll
  for (int ct = 0; ct < 4; ++ct) {
    const int cg = c0 + cwBase + ct * 16 + lo;
    const float bb = bo[cg];
#pragma unroll
    for (int v = 0; v < 8; ++v) {
      const int rg = r0 + rw + v + hi * 8;
      out[(size_t)rg * Cc + cg] = acc[ct][v] + bb;
    }
  }
}

// ---------------------------------------------------------------------------
extern "C" void kernel_launch(void* const* d_in, const int* in_sizes, int n_in,
                              void* d_out, int out_size, void* d_ws,
                              size_t ws_size, hipStream_t stream) {
  (void)in_sizes; (void)n_in; (void)out_size; (void)ws_size;
  const float* x  = (const float*)d_in[0];
  const float* Wq = (const float*)d_in[1];
  const float* bq = (const float*)d_in[2];
  const float* Wk = (const float*)d_in[3];
  const float* bk = (const float*)d_in[4];
  const float* Wv = (const float*)d_in[5];
  const float* bv = (const float*)d_in[6];
  const float* Wo = (const float*)d_in[7];
  const float* bo = (const float*)d_in[8];
  float* out = (float*)d_out;

  const size_t per = (size_t)Bb * Hh * Nn * Dd;  // 33,554,432 bf16 elems
  u16* Qh   = (u16*)d_ws;
  u16* Kh   = Qh + per;
  u16* Vh   = Kh + per;
  u16* attn = Vh + per;

  dim3 g1(Mrows / 64, Cc / 128);
  qkv_proj_kernel<<<g1, 256, 0, stream>>>(x, Wq, bq, Wk, bk, Wv, bv, Qh, Kh, Vh);

  dim3 g2(Bb * Hh, 2);
  attn_kernel<<<g2, 256, 0, stream>>>(Qh, Kh, Vh, attn);

  dim3 g3(Mrows / 64, Cc / 128);
  out_proj_kernel<<<g3, 256, 0, stream>>>(attn, Wo, bo, out);
}
